// GraphConvLayer_1297080123525
// MI455X (gfx1250) — matrix operand order repaired and checked by reading, not compiled, not run on hardware
//
#include <hip/hip_runtime.h>
#include <hip/hip_bf16.h>
#include <math.h>

#define N_NODES 50000
#define N_EDGES 800000
#define IN_CH 64
#define OUT_CH 128
#define N_HEADS 4
#define HEAD_DIM 32
#define QKV_W 384
#define LN_EPS 1e-5f
#define INV_SQRT_HD 0.17677669529663687f

typedef _Float16 h16;
typedef __attribute__((ext_vector_type(16))) _Float16 v16h;
typedef __attribute__((ext_vector_type(8)))  _Float16 v8h;
typedef __attribute__((ext_vector_type(8)))  float    v8f;

static __device__ inline v8f wmma16x16x32(v16h a, v16h b, v8f c) {
  // D = A(16x32 f16) x B(32x16 f16) + C(16x16 f32)
  return __builtin_amdgcn_wmma_f32_16x16x32_f16(false, a, false, b, (short)0, c, false, false);
}

// order-preserving float <-> uint mapping for atomicMax on floats
static __device__ inline unsigned f32_ordered(float f) {
  unsigned u = __float_as_uint(f);
  return (u & 0x80000000u) ? ~u : (u | 0x80000000u);
}
static __device__ inline float f32_unordered(unsigned u) {
  unsigned v = (u & 0x80000000u) ? (u & 0x7FFFFFFFu) : ~u;
  return __uint_as_float(v);
}

// ---------------- init: zero aggregator, reset max/sum ----------------
__global__ void k_init(float* __restrict__ agg, unsigned* __restrict__ maxv,
                       float* __restrict__ sumv) {
  long i = (long)blockIdx.x * blockDim.x + threadIdx.x;
  const long n = (long)N_NODES * OUT_CH;
  for (long j = i; j < n; j += (long)gridDim.x * blockDim.x) agg[j] = 0.0f;
  if (i < N_HEADS) { maxv[i] = 0u; sumv[i] = 0.0f; }
}

// ---------------- fp32 -> fp16 conversion of x ----------------
__global__ void k_convert_x(const float* __restrict__ x, h16* __restrict__ xh) {
  int i = blockIdx.x * blockDim.x + threadIdx.x;  // exactly N_NODES*IN_CH threads
  xh[i] = (h16)x[i];
}

// ------------- weights -> fp16, PRE-PACKED per-lane WMMA B-fragment layout -------------
// Packed layout for W (K x N row-major): dst[((ct*numKK + kk)*32 + lane)*16 + idx] =
//   W[(kk*32 + (idx>>3)*16 + (lane>=16 ? 8 : 0) + (idx&7)) * N + ct*16 + (lane & 15)]
// so each lane's B fragment is 16 contiguous halfs (one 32-byte load).
__global__ void k_convert_w(const float* __restrict__ Wq, const float* __restrict__ bq,
                            const float* __restrict__ Wk, const float* __restrict__ bk,
                            const float* __restrict__ Wv, const float* __restrict__ bv,
                            const float* __restrict__ We1, const float* __restrict__ Wo,
                            h16* __restrict__ wqkvP, float* __restrict__ bqkv,
                            h16* __restrict__ we1P, h16* __restrict__ woP) {
  int i = blockIdx.x * blockDim.x + threadIdx.x;  // 64*384 = 24576 threads
  {
    // qkv weights: K=64 (numKK=2), N=384 (24 col tiles)
    const int idx  = i & 15;
    const int lane = (i >> 4) & 31;
    const int frag = i >> 9;            // 0..47
    const int kk = frag & 1;
    const int ct = frag >> 1;           // 0..23
    const int k = kk * 32 + (idx >> 3) * 16 + ((lane >= 16) ? 8 : 0) + (idx & 7);
    const int c = ct * 16 + (lane & 15);
    float v = (c < 128) ? Wq[k * 128 + c]
            : (c < 256) ? Wk[k * 128 + (c - 128)]
                        : Wv[k * 128 + (c - 256)];
    wqkvP[i] = (h16)v;
  }
  if (i < QKV_W)
    bqkv[i] = (i < 128) ? bq[i] : (i < 256) ? bk[i - 128] : bv[i - 256];
  if (i < OUT_CH * OUT_CH) {
    // 128x128 weights: K=128 (numKK=4), N=128 (8 col tiles)
    const int idx  = i & 15;
    const int lane = (i >> 4) & 31;
    const int frag = i >> 9;            // 0..31
    const int kk = frag & 3;
    const int ct = frag >> 2;           // 0..7
    const int k = kk * 32 + (idx >> 3) * 16 + ((lane >= 16) ? 8 : 0) + (idx & 7);
    const int c = ct * 16 + (lane & 15);
    we1P[i] = (h16)We1[k * OUT_CH + c];
    woP[i]  = (h16)Wo[k * OUT_CH + c];
  }
}

// ---------------- qkv = x @ [Wq|Wk|Wv] + b : 50000x64 @ 64x384 ----------------
__global__ void __launch_bounds__(256) k_qkv(const h16* __restrict__ xh,
                                             const h16* __restrict__ wqkvP,
                                             const float* __restrict__ bqkv,
                                             float* __restrict__ qkv) {
  const int lane = threadIdx.x & 31;
  const int wave = threadIdx.x >> 5;      // 0..7
  const int rt = blockIdx.x;              // 0..3124 (16-row tiles)
  const int rlo = lane & 15;
  const int hi16 = (lane >= 16) ? 1 : 0;

  __builtin_prefetch(wqkvP + (size_t)threadIdx.x * 64, 0, 1);

  // A fragments: row rt*16+rlo, K=0..31 and K=32..63 (ISA 16-bit A 16x32 layout)
  v16h a[2];
  const h16* xrow = xh + (size_t)(rt * 16 + rlo) * IN_CH;
#pragma unroll
  for (int kk = 0; kk < 2; ++kk) {
    v8h lo = *(const v8h*)(xrow + kk * 32 + hi16 * 8);
    v8h hi = *(const v8h*)(xrow + kk * 32 + 16 + hi16 * 8);
#pragma unroll
    for (int i = 0; i < 8; ++i) { a[kk][i] = lo[i]; a[kk][i + 8] = hi[i]; }
  }

#pragma unroll
  for (int t = 0; t < 3; ++t) {
    const int ct = wave * 3 + t;          // 24 col tiles
    const int col = ct * 16 + rlo;
    v8f acc = {};
#pragma unroll
    for (int kk = 0; kk < 2; ++kk) {
      v16h b = *(const v16h*)(wqkvP + (size_t)((ct * 2 + kk) * 32 + lane) * 16);
      acc = wmma16x16x32(a[kk], b, acc);
    }
    const float bias = bqkv[col];
#pragma unroll
    for (int r = 0; r < 8; ++r) {
      const int row = rt * 16 + r + hi16 * 8;
      qkv[(size_t)row * QKV_W + col] = acc[r] + bias;
    }
  }
}

// -------- edge MLP + attention logits: 64 edges per block, 8 waves --------
__global__ void __launch_bounds__(256) k_edge(const h16* __restrict__ xh,
                                              const int* __restrict__ eidx,
                                              const h16* __restrict__ we1P,
                                              const float* __restrict__ be1,
                                              const float* __restrict__ We2,
                                              const float* __restrict__ be2,
                                              const float* __restrict__ qkv,
                                              float* __restrict__ logits,
                                              unsigned* __restrict__ maxv) {
  __shared__ h16 Alds[64 * 128];        // 16 KB  gathered edge features (f16)
  __shared__ float Hlds[64 * 128];      // 32 KB  relu hidden layer (f32)
  __shared__ float W2lds[4 * 128];      //  2 KB  We2 transposed [head][128]
  __shared__ unsigned redmax[4];

  const int t = threadIdx.x;
  const int e0 = blockIdx.x * 64;
  const int* srcIdx = eidx;
  const int* dstIdx = eidx + N_EDGES;

  __builtin_prefetch(we1P + (size_t)t * 64, 0, 1);

  // stage A: edge el features = [x[src] (64), x[dst] (64)] in f16
  {
    const int el = t >> 2, part = t & 3;
    const int e = e0 + el;
    const int node = (part < 2) ? srcIdx[e] : dstIdx[e];
    const int off = (part & 1) * 32;
    const v8h* sp = (const v8h*)(xh + (size_t)node * IN_CH + off);
    v8h* dp = (v8h*)(Alds + el * 128 + ((part >= 2) ? 64 : 0) + off);
#pragma unroll
    for (int i = 0; i < 4; ++i) dp[i] = sp[i];
  }
  // stage We2 transposed: We2 is [128][4] row-major
  {
    W2lds[(t & 3) * 128 + (t >> 2)] = We2[t];
    const int t2 = t + 256;
    W2lds[(t2 & 3) * 128 + (t2 >> 2)] = We2[t2];
  }
  if (t < 4) redmax[t] = 0u;
  __syncthreads();

  // WMMA: H(64x128) = A(64x128) @ We1(128x128); wave -> (mtile, 4 ntiles)
  const int lane = t & 31, wave = t >> 5;
  const int m = wave & 3, g = wave >> 2;
  const int rlo = lane & 15;
  const int hi16 = (lane >= 16) ? 1 : 0;
  v8f acc[4] = {{}, {}, {}, {}};
  for (int kk = 0; kk < 4; ++kk) {
    v16h a;
    const h16* ar = Alds + (m * 16 + rlo) * 128 + kk * 32 + hi16 * 8;
    {
      v8h lo = *(const v8h*)(ar);
      v8h hi = *(const v8h*)(ar + 16);
#pragma unroll
      for (int i = 0; i < 8; ++i) { a[i] = lo[i]; a[i + 8] = hi[i]; }
    }
#pragma unroll
    for (int nt = 0; nt < 4; ++nt) {
      v16h b = *(const v16h*)(we1P + (size_t)(((g * 4 + nt) * 4 + kk) * 32 + lane) * 16);
      acc[nt] = wmma16x16x32(a, b, acc[nt]);
    }
  }
  // bias + relu -> Hlds
#pragma unroll
  for (int nt = 0; nt < 4; ++nt) {
    const int col = (g * 4 + nt) * 16 + rlo;
    const float bias = be1[col];
#pragma unroll
    for (int r = 0; r < 8; ++r) {
      float hv = acc[nt][r] + bias;
      Hlds[(m * 16 + r + hi16 * 8) * 128 + col] = hv > 0.0f ? hv : 0.0f;
    }
  }
  __syncthreads();

  // phase 2: per (edge, head): 128->4 projection, q.k dot, logit
  {
    const int el = t >> 2, hh = t & 3;
    const int e = e0 + el;
    float w = be2[hh];
#pragma unroll 8
    for (int j = 0; j < 128; ++j) w += Hlds[el * 128 + j] * W2lds[hh * 128 + j];
    const int s = srcIdx[e], d = dstIdx[e];
    const float* qrow = qkv + (size_t)d * QKV_W + hh * HEAD_DIM;         // q
    const float* krow = qkv + (size_t)s * QKV_W + 128 + hh * HEAD_DIM;   // k
    float dot = 0.0f;
#pragma unroll
    for (int i = 0; i < HEAD_DIM; ++i) dot += qrow[i] * krow[i];
    const float logit = dot * w * INV_SQRT_HD;
    logits[(size_t)e * 4 + hh] = logit;
    atomicMax(&redmax[hh], f32_ordered(logit));
  }
  __syncthreads();
  if (t < 4) atomicMax(&maxv[t], redmax[t]);
}

// ---------------- global softmax denominator ----------------
__global__ void k_expsum(const float* __restrict__ logits,
                         const unsigned* __restrict__ maxv,
                         float* __restrict__ sumv) {
  __shared__ float red[4];
  if (threadIdx.x < 4) red[threadIdx.x] = 0.0f;
  __syncthreads();
  float mx[4];
#pragma unroll
  for (int h = 0; h < 4; ++h) mx[h] = f32_unordered(maxv[h]);
  const long n = (long)N_EDGES * 4;
  float part[4] = {0.f, 0.f, 0.f, 0.f};
  for (long i = (long)blockIdx.x * blockDim.x + threadIdx.x; i < n;
       i += (long)gridDim.x * blockDim.x) {
    const int h = (int)(i & 3);
    part[h] += expf(logits[i] - mx[h]);
  }
#pragma unroll
  for (int h = 0; h < 4; ++h)
    if (part[h] != 0.0f) atomicAdd(&red[h], part[h]);
  __syncthreads();
  if (threadIdx.x < 4) atomicAdd(&sumv[threadIdx.x], red[threadIdx.x]);
}

// ---------------- scatter: agg[dst] += softmax(attn) * v[src] ----------------
__global__ void __launch_bounds__(256) k_scatter(const float* __restrict__ logits,
                                                 const unsigned* __restrict__ maxv,
                                                 const float* __restrict__ sumv,
                                                 const float* __restrict__ qkv,
                                                 const int* __restrict__ eidx,
                                                 float* __restrict__ agg) {
  __shared__ float smax[4], sinv[4];
  if (threadIdx.x < 4) {
    smax[threadIdx.x] = f32_unordered(maxv[threadIdx.x]);
    sinv[threadIdx.x] = 1.0f / sumv[threadIdx.x];
  }
  __syncthreads();
  const int e = blockIdx.x * 2 + (threadIdx.x >> 7);
  const int ch = threadIdx.x & 127;
  const int hh = ch >> 5;
  const int s = eidx[e];
  const int d = eidx[e + N_EDGES];
  const float l = logits[(size_t)e * 4 + hh];
  const float coef = expf(l - smax[hh]) * sinv[hh];
  const float v = qkv[(size_t)s * QKV_W + 256 + ch];
  atomicAdd(&agg[(size_t)d * OUT_CH + ch], coef * v);
}

// ---------------- out = LN(agg @ Wo + bo) ----------------
__global__ void __launch_bounds__(256) k_out(const float* __restrict__ agg,
                                             const h16* __restrict__ woP,
                                             const float* __restrict__ bo,
                                             const float* __restrict__ gamma,
                                             const float* __restrict__ beta,
                                             float* __restrict__ out) {
  __shared__ float O[16 * 128];
  __shared__ float RS[16 * 16];
  __shared__ float RSS[16 * 16];
  __shared__ float MU[16], RV[16];
  const int t = threadIdx.x, lane = t & 31, wave = t >> 5;
  const int rt = blockIdx.x;                 // 16-row tile
  const int rlo = lane & 15;
  const int hi16 = (lane >= 16) ? 1 : 0;
  const int col = wave * 16 + rlo;           // 8 waves cover 128 cols

  __builtin_prefetch(woP + (size_t)t * 64, 0, 1);

  const float* arow = agg + (size_t)(rt * 16 + rlo) * OUT_CH;
  v8f acc = {};
  for (int kk = 0; kk < 4; ++kk) {
    v16h a;
    const float* p0 = arow + kk * 32 + hi16 * 8;
    const float* p1 = arow + kk * 32 + 16 + hi16 * 8;
#pragma unroll
    for (int i = 0; i < 8; ++i) { a[i] = (h16)p0[i]; a[i + 8] = (h16)p1[i]; }
    v16h b = *(const v16h*)(woP + (size_t)((wave * 4 + kk) * 32 + lane) * 16);
    acc = wmma16x16x32(a, b, acc);
  }
  const float bias = bo[col];
#pragma unroll
  for (int r = 0; r < 8; ++r) O[(r + hi16 * 8) * 128 + col] = acc[r] + bias;
  __syncthreads();

  // LayerNorm over 128 cols per row (16 rows, 16 threads/row, 8 cols/thread)
  const int r = t >> 4, j = t & 15;
  float s = 0.0f, ss = 0.0f;
#pragma unroll
  for (int c = j * 8; c < j * 8 + 8; ++c) { float v = O[r * 128 + c]; s += v; ss += v * v; }
  RS[r * 16 + j] = s; RSS[r * 16 + j] = ss;
  __syncthreads();
  if (j == 0) {
    float S = 0.0f, SS = 0.0f;
#pragma unroll
    for (int q = 0; q < 16; ++q) { S += RS[r * 16 + q]; SS += RSS[r * 16 + q]; }
    const float mu = S * (1.0f / 128.0f);
    const float var = SS * (1.0f / 128.0f) - mu * mu;
    MU[r] = mu; RV[r] = rsqrtf(var + LN_EPS);
  }
  __syncthreads();
  const float mu = MU[r], rv = RV[r];
  const size_t row = (size_t)(rt * 16 + r) * OUT_CH;
#pragma unroll
  for (int c = j * 8; c < j * 8 + 8; ++c)
    out[row + c] = (O[r * 128 + c] - mu) * rv * gamma[c] + beta[c];
}

extern "C" void kernel_launch(void* const* d_in, const int* in_sizes, int n_in,
                              void* d_out, int out_size, void* d_ws, size_t ws_size,
                              hipStream_t stream) {
  (void)in_sizes; (void)n_in; (void)out_size; (void)ws_size;
  const float* x    = (const float*)d_in[0];
  const int*   eidx = (const int*)d_in[1];
  const float* Wq = (const float*)d_in[2];  const float* bq = (const float*)d_in[3];
  const float* Wk = (const float*)d_in[4];  const float* bk = (const float*)d_in[5];
  const float* Wv = (const float*)d_in[6];  const float* bv = (const float*)d_in[7];
  const float* We1 = (const float*)d_in[8]; const float* be1 = (const float*)d_in[9];
  const float* We2 = (const float*)d_in[10]; const float* be2 = (const float*)d_in[11];
  const float* Wo = (const float*)d_in[12]; const float* bo = (const float*)d_in[13];
  const float* gamma = (const float*)d_in[14]; const float* beta = (const float*)d_in[15];
  float* out = (float*)d_out;

  char* ws = (char*)d_ws;
  size_t off = 0;
  auto carve = [&](size_t bytes) {
    char* p = ws + off;
    off += (bytes + 255) & ~(size_t)255;
    return p;
  };
  h16*      xh     = (h16*)     carve(sizeof(h16) * (size_t)N_NODES * IN_CH);
  h16*      wqkvP  = (h16*)     carve(sizeof(h16) * 64 * QKV_W);
  float*    bqkv   = (float*)   carve(sizeof(float) * QKV_W);
  h16*      we1P   = (h16*)     carve(sizeof(h16) * OUT_CH * OUT_CH);
  h16*      woP    = (h16*)     carve(sizeof(h16) * OUT_CH * OUT_CH);
  unsigned* maxv   = (unsigned*)carve(sizeof(unsigned) * N_HEADS);
  float*    sumv   = (float*)   carve(sizeof(float) * N_HEADS);
  float*    qkv    = (float*)   carve(sizeof(float) * (size_t)N_NODES * QKV_W);
  float*    logits = (float*)   carve(sizeof(float) * (size_t)N_EDGES * N_HEADS);
  float*    agg    = (float*)   carve(sizeof(float) * (size_t)N_NODES * OUT_CH);

  k_init<<<4096, 256, 0, stream>>>(agg, maxv, sumv);
  k_convert_x<<<(N_NODES * IN_CH) / 256, 256, 0, stream>>>(x, xh);
  k_convert_w<<<(64 * QKV_W) / 256, 256, 0, stream>>>(Wq, bq, Wk, bk, Wv, bv, We1, Wo,
                                                      wqkvP, bqkv, we1P, woP);
  k_qkv<<<N_NODES / 16, 256, 0, stream>>>(xh, wqkvP, bqkv, qkv);
  k_edge<<<N_EDGES / 64, 256, 0, stream>>>(xh, eidx, we1P, be1, We2, be2, qkv, logits, maxv);
  k_expsum<<<1024, 256, 0, stream>>>(logits, maxv, sumv);
  k_scatter<<<N_EDGES / 2, 256, 0, stream>>>(logits, maxv, sumv, qkv, eidx, agg);
  k_out<<<N_NODES / 16, 256, 0, stream>>>(agg, woP, bo, gamma, beta, out);
}
